// SwinHuggingFaceDQN_24644522345352
// MI455X (gfx1250) — matrix operand order repaired and checked by reading, not compiled, not run on hardware
//
#include <hip/hip_runtime.h>
#include <hip/hip_bf16.h>
#include <math.h>

typedef __attribute__((ext_vector_type(16))) _Float16 v16h;
typedef __attribute__((ext_vector_type(8)))  _Float16 v8h;
typedef __attribute__((ext_vector_type(4)))  _Float16 v4h;
typedef __attribute__((ext_vector_type(8)))  float    v8f;

__device__ __forceinline__ float gelu_exact(float x) {
    return 0.5f * x * (1.0f + erff(x * 0.70710678118654752f));
}

// Build a v16h A-fragment from an LDS row (lanes 0-15: K{0..15},
// lanes 16-31: K{16..31} handled by caller-provided kb offset).
__device__ __forceinline__ v16h make_afrag(const _Float16* rowp, int kb) {
    v8h alo = *(const v8h*)(rowp + (kb & ~15) + ((kb & 8) << 1));
    v8h ahi = *(const v8h*)(rowp + (kb & ~15) + ((kb & 8) << 1) + 8);
    v16h a;
#pragma unroll
    for (int i = 0; i < 8; ++i) { a[i] = alo[i]; a[i + 8] = ahi[i]; }
    return a;
}

// ---------------------------------------------------------------------------
// FAST GEMM: C[M,N] = act(A[M,K] @ B[K,N] + bias) + R
// Requirements: M % 128 == 0, K % 32 == 0, N % 4 == 0, B row-major [K,N].
// 256 threads = 8 waves; tile 128x64; K-step 32; double-buffered LDS staging
// (global fetch of step k+1 overlaps the WMMAs of step k).
// ---------------------------------------------------------------------------
__global__ __launch_bounds__(256)
void gemm_f16wmma_fast(const float* __restrict__ A, const float* __restrict__ Bm,
                       const float* __restrict__ bias, const float* R,
                       float* C, int M, int N, int K, int act)
{
    __shared__ alignas(64) _Float16 Ah[2][128][32];  // [buf][m][k]
    __shared__ alignas(64) _Float16 Bh[2][64][32];   // [buf][n][k]
    const int t    = threadIdx.x;
    const int wv   = t >> 5;
    const int lane = t & 31;
    const int m0   = blockIdx.y * 128;
    const int n0   = blockIdx.x * 64;

    // precomputed staging coordinates
    int aR[4], aC[4];
#pragma unroll
    for (int i = 0; i < 4; ++i) {
        int lin4 = t + i * 256;            // 1024 float4 groups of A (128x32)
        aR[i] = lin4 >> 3;
        aC[i] = (lin4 & 7) * 4;
    }
    int bK[2], bNg[2], bNl[2];
    bool bOK[2];
#pragma unroll
    for (int i = 0; i < 2; ++i) {
        int lin4 = t + i * 256;            // 512 float4 groups of B (32x64)
        bK[i]  = lin4 >> 4;
        int nl = (lin4 & 15) * 4;
        bNl[i] = nl;
        int gn = n0 + nl;
        bOK[i] = gn < N;                   // N%4==0 -> float4 fully in/out
        bNg[i] = bOK[i] ? gn : 0;
    }

    float4 ra[4], rb[2];
    auto fetch = [&](int kk) {
#pragma unroll
        for (int i = 0; i < 4; ++i)
            ra[i] = *(const float4*)&A[(size_t)(m0 + aR[i]) * K + kk + aC[i]];
#pragma unroll
        for (int i = 0; i < 2; ++i)
            rb[i] = *(const float4*)&Bm[(size_t)(kk + bK[i]) * N + bNg[i]];
    };
    auto stage = [&](int buf) {
#pragma unroll
        for (int i = 0; i < 4; ++i) {
            v4h hv;
            hv[0] = (_Float16)ra[i].x; hv[1] = (_Float16)ra[i].y;
            hv[2] = (_Float16)ra[i].z; hv[3] = (_Float16)ra[i].w;
            *(v4h*)&Ah[buf][aR[i]][aC[i]] = hv;
        }
#pragma unroll
        for (int i = 0; i < 2; ++i) {
            Bh[buf][bNl[i] + 0][bK[i]] = (_Float16)(bOK[i] ? rb[i].x : 0.0f);
            Bh[buf][bNl[i] + 1][bK[i]] = (_Float16)(bOK[i] ? rb[i].y : 0.0f);
            Bh[buf][bNl[i] + 2][bK[i]] = (_Float16)(bOK[i] ? rb[i].z : 0.0f);
            Bh[buf][bNl[i] + 3][bK[i]] = (_Float16)(bOK[i] ? rb[i].w : 0.0f);
        }
    };

    v8f acc[4];
    for (int i = 0; i < 4; ++i)
        for (int j = 0; j < 8; ++j) acc[i][j] = 0.0f;

    fetch(0);
    stage(0);
    int cur = 0;
    for (int kk = 0; kk < K; kk += 32) {
        __syncthreads();
        bool nxt = (kk + 32) < K;
        if (nxt) fetch(kk + 32);           // overlap global loads with WMMAs

        const int r   = (wv << 4) + (lane & 15);
        const int kb  = (lane >> 4) << 3;
        const v16h a  = make_afrag(&Ah[cur][r][0], kb);
        const int kb2 = (lane >> 4) << 4;
#pragma unroll
        for (int nt = 0; nt < 4; ++nt) {
            const v16h b = *(const v16h*)&Bh[cur][(nt << 4) + (lane & 15)][kb2];
            acc[nt] = __builtin_amdgcn_wmma_f32_16x16x32_f16(
                false, a, false, b, (short)0, acc[nt], false, false);
        }
        __syncthreads();
        if (nxt) { stage(cur ^ 1); cur ^= 1; }
    }

    const int mrow  = m0 + (wv << 4) + ((lane >> 4) << 3);
    const int ncol0 = n0 + (lane & 15);
#pragma unroll
    for (int nt = 0; nt < 4; ++nt) {
        int n = ncol0 + (nt << 4);
        if (n >= N) continue;
        float bv = bias ? bias[n] : 0.0f;
#pragma unroll
        for (int g = 0; g < 8; ++g) {
            int m = mrow + g;
            float v = acc[nt][g] + bv;
            if (act == 1) v = gelu_exact(v);
            if (R) v += R[(size_t)m * N + n];
            C[(size_t)m * N + n] = v;
        }
    }
}

// ---------------------------------------------------------------------------
// SAFE GEMM (any M,N,K; bTrans for [O,K] weights). 128 threads, 64x64 tile.
// Clamped unconditional loads (no divergent guards in the hot loop).
// ---------------------------------------------------------------------------
__global__ __launch_bounds__(128)
void gemm_f16wmma_safe(const float* __restrict__ A, const float* __restrict__ Bm,
                       const float* __restrict__ bias, const float* R,
                       float* C, int M, int N, int K, int bTrans, int act)
{
    __shared__ alignas(64) _Float16 Ah[64][32];
    __shared__ alignas(64) _Float16 Bh[64][32];
    const int t    = threadIdx.x;
    const int wv   = t >> 5;
    const int lane = t & 31;
    const int m0   = blockIdx.y * 64;
    const int n0   = blockIdx.x * 64;

    v8f acc[4];
    for (int i = 0; i < 4; ++i)
        for (int j = 0; j < 8; ++j) acc[i][j] = 0.0f;

    for (int kk = 0; kk < K; kk += 32) {
#pragma unroll
        for (int i = 0; i < 16; ++i) {
            int lin = t + i * 128;
            int r = lin >> 5, k = lin & 31;
            int gm = m0 + r, gk = kk + k;
            int sm = gm < M ? gm : M - 1;
            int sk = gk < K ? gk : K - 1;
            float v = A[(size_t)sm * K + sk];
            v = (gm < M && gk < K) ? v : 0.0f;
            Ah[r][k] = (_Float16)v;
        }
#pragma unroll
        for (int i = 0; i < 16; ++i) {
            int lin = t + i * 128;
            int k = lin >> 6, n = lin & 63;
            int gk = kk + k, gn = n0 + n;
            int sk = gk < K ? gk : K - 1;
            int sn = gn < N ? gn : N - 1;
            float v = bTrans ? Bm[(size_t)sn * K + sk] : Bm[(size_t)sk * N + sn];
            v = (gk < K && gn < N) ? v : 0.0f;
            Bh[n][k] = (_Float16)v;
        }
        __syncthreads();

        const int r   = (wv << 4) + (lane & 15);
        const int kb  = (lane >> 4) << 3;
        const v16h a  = make_afrag(&Ah[r][0], kb);
        const int kb2 = (lane >> 4) << 4;
#pragma unroll
        for (int nt = 0; nt < 4; ++nt) {
            const v16h b = *(const v16h*)&Bh[(nt << 4) + (lane & 15)][kb2];
            acc[nt] = __builtin_amdgcn_wmma_f32_16x16x32_f16(
                false, a, false, b, (short)0, acc[nt], false, false);
        }
        __syncthreads();
    }

    const int mrow  = m0 + (wv << 4) + ((lane >> 4) << 3);
    const int ncol0 = n0 + (lane & 15);
#pragma unroll
    for (int nt = 0; nt < 4; ++nt) {
        int n = ncol0 + (nt << 4);
        if (n >= N) continue;
        float bv = bias ? bias[n] : 0.0f;
#pragma unroll
        for (int g = 0; g < 8; ++g) {
            int m = mrow + g;
            if (m >= M) continue;
            float v = acc[nt][g] + bv;
            if (act == 1) v = gelu_exact(v);
            if (R) v += R[(size_t)m * N + n];
            C[(size_t)m * N + n] = v;
        }
    }
}

// ---------------------------------------------------------------------------
// LayerNorm: one wave32 per row
// ---------------------------------------------------------------------------
__global__ __launch_bounds__(256)
void layernorm_k(const float* __restrict__ x, const float* __restrict__ g,
                 const float* __restrict__ b, float* __restrict__ y,
                 int rows, int C)
{
    int wv = threadIdx.x >> 5, lane = threadIdx.x & 31;
    int row = blockIdx.x * 8 + wv;
    if (row >= rows) return;
    const float* xr = x + (size_t)row * C;
    float s = 0.0f, sq = 0.0f;
    for (int c = lane; c < C; c += 32) { float v = xr[c]; s += v; sq += v * v; }
    for (int off = 16; off > 0; off >>= 1) {
        s  += __shfl_xor(s, off, 32);
        sq += __shfl_xor(sq, off, 32);
    }
    float mean = s / C;
    float var  = sq / C - mean * mean;
    float rstd = rsqrtf(var + 1e-5f);
    float* yr = y + (size_t)row * C;
    for (int c = lane; c < C; c += 32)
        yr[c] = (xr[c] - mean) * rstd * g[c] + b[c];
}

// ---------------------------------------------------------------------------
// im2col for 3x3/stride-3 patch embed, with /255 scaling.
// ---------------------------------------------------------------------------
__global__ void im2col_patch(const float* __restrict__ X, float* __restrict__ A, int Bn)
{
    int idx = blockIdx.x * 256 + threadIdx.x;
    int total = Bn * 784 * 36;
    if (idx >= total) return;
    int k = idx % 36, m = idx / 36;
    int b = m / 784, p = m % 784;
    int ph = p / 28, pw = p % 28;
    int c = k / 9, rem = k % 9;
    int r3 = rem / 3, c3 = rem % 3;
    A[idx] = X[(((size_t)b * 4 + c) * 84 + (ph * 3 + r3)) * 84 + (pw * 3 + c3)] * (1.0f / 255.0f);
}

// ---------------------------------------------------------------------------
// Window partition with cyclic shift: x [B, H*H, C] -> o [B*nWin*49, C]
// ---------------------------------------------------------------------------
__global__ void win_part(const float* __restrict__ x, float* __restrict__ o,
                         int Bn, int H, int C, int shift)
{
    int idx = blockIdx.x * 256 + threadIdx.x;
    int total = Bn * H * H * C;
    if (idx >= total) return;
    int c = idx % C, t = idx / C;
    int b = t / (H * H), rr = t % (H * H);
    int win = rr / 49, pos = rr % 49;
    int nW = H / 7;
    int wh = win / nW, ww = win % nW;
    int i = pos / 7, j = pos % 7;
    int r  = (wh * 7 + i + shift) % H;
    int cl = (ww * 7 + j + shift) % H;
    o[(size_t)t * C + c] = x[((size_t)b * H * H + r * H + cl) * C + c];
}

// ---------------------------------------------------------------------------
// Window reverse + un-shift + residual add (in place into h)
// ---------------------------------------------------------------------------
__global__ void win_rev_add(float* h, const float* __restrict__ pw,
                            int Bn, int H, int C, int shift)
{
    int idx = blockIdx.x * 256 + threadIdx.x;
    int total = Bn * H * H * C;
    if (idx >= total) return;
    int c = idx % C, t = idx / C;
    int b = t / (H * H), rr = t % (H * H);
    int r = rr / H, cl = rr % H;
    int u = (r - shift + H) % H;
    int v = (cl - shift + H) % H;
    int nW = H / 7;
    int wh = u / 7, i = u % 7, ww = v / 7, j = v % 7;
    int srow = (b * nW * nW + wh * nW + ww) * 49 + i * 7 + j;
    h[(size_t)t * C + c] += pw[(size_t)srow * C + c];
}

// ---------------------------------------------------------------------------
// Fused WMMA window attention: one block (128 threads = 4 waves) per
// (window, head). Q/K/V zero-padded to 64x64 f16 tiles in LDS.
// S = Q.K^T (WMMA) + scale + rel-bias + shift-mask, softmax, O = P.V (WMMA).
// ---------------------------------------------------------------------------
__global__ __launch_bounds__(128)
void win_attn(const float* __restrict__ Q, const float* __restrict__ K,
              const float* __restrict__ V, const float* __restrict__ bias_tbl,
              float* __restrict__ O, int heads, int d, int C, int H, int shift)
{
    __shared__ alignas(64) _Float16 qh[64][64];  // [token][k]
    __shared__ alignas(64) _Float16 kh[64][64];  // [token][k]  (B-layout of K^T)
    __shared__ alignas(64) _Float16 vt[64][64];  // [channel][token] (B-layout of V)
    __shared__ alignas(64) float    sf[64][64];
    __shared__ alignas(64) _Float16 ph[64][64];
    const int w  = blockIdx.x / heads;
    const int hd = blockIdx.x % heads;
    const int t  = threadIdx.x;
    const int wv = t >> 5, lane = t & 31;
    const int nW = H / 7, nWin = nW * nW;

    // zero-fill (padding regions)
    {
        v4h z;
        z[0] = z[1] = z[2] = z[3] = (_Float16)0.0f;
        for (int idx = t; idx < 64 * 64 / 4; idx += 128) {
            ((v4h*)qh)[idx] = z;
            ((v4h*)kh)[idx] = z;
            ((v4h*)vt)[idx] = z;
        }
    }
    __syncthreads();
    for (int idx = t; idx < 49 * d; idx += 128) {
        int i = idx / d, kk = idx % d;
        size_t row = (size_t)(w * 49 + i) * C + hd * d + kk;
        qh[i][kk] = (_Float16)Q[row];
        kh[i][kk] = (_Float16)K[row];
        vt[kk][i] = (_Float16)V[row];
    }
    __syncthreads();

    const int rloc = (wv << 4) + (lane & 15);
    const int kbo  = (lane >> 4) << 3;
    const int kbo2 = (lane >> 4) << 4;
    const int mrow = (wv << 4) + ((lane >> 4) << 3);
    const int ncol = lane & 15;

    // ---- S = Q @ K^T ----
    v8f acc[4];
    for (int i = 0; i < 4; ++i)
        for (int j = 0; j < 8; ++j) acc[i][j] = 0.0f;
    for (int kk = 0; kk < d; kk += 32) {
        const v16h a = make_afrag(&qh[rloc][0], kk + kbo);
#pragma unroll
        for (int nt = 0; nt < 4; ++nt) {
            const v16h b = *(const v16h*)&kh[(nt << 4) + (lane & 15)][kk + kbo2];
            acc[nt] = __builtin_amdgcn_wmma_f32_16x16x32_f16(
                false, a, false, b, (short)0, acc[nt], false, false);
        }
    }

    // epilogue: scale + relative-position bias + shift mask into sf (f32)
    const float scale = rsqrtf((float)d);
    const int wl  = w % nWin;
    const int whr = (wl / nW) * 7, wwc = (wl % nW) * 7;
#pragma unroll
    for (int nt = 0; nt < 4; ++nt) {
        int n = ncol + (nt << 4);
#pragma unroll
        for (int g = 0; g < 8; ++g) {
            int m = mrow + g;
            float v = -1e9f;
            if (m < 49 && n < 49) {
                v = acc[nt][g] * scale;
                int ri = m / 7, ci = m % 7, rj = n / 7, cj = n % 7;
                v += bias_tbl[((ri - rj + 6) * 13 + (ci - cj + 6)) * heads + hd];
                if (shift > 0) {
                    int gri = whr + ri, gci = wwc + ci, grj = whr + rj, gcj = wwc + cj;
                    int ra = (gri < H - 7) ? 0 : ((gri < H - shift) ? 1 : 2);
                    int ca = (gci < H - 7) ? 0 : ((gci < H - shift) ? 1 : 2);
                    int rb = (grj < H - 7) ? 0 : ((grj < H - shift) ? 1 : 2);
                    int cb = (gcj < H - 7) ? 0 : ((gcj < H - shift) ? 1 : 2);
                    if (ra * 3 + ca != rb * 3 + cb) v += -100.0f;
                }
            }
            sf[m][n] = v;
        }
    }
    __syncthreads();

    // softmax over valid rows (padded cols hold -1e9 -> exp == 0)
    if (t < 49) {
        float mx = -1e30f;
        for (int j = 0; j < 64; ++j) mx = fmaxf(mx, sf[t][j]);
        float sum = 0.0f;
        for (int j = 0; j < 64; ++j) {
            float e = __expf(sf[t][j] - mx);
            sf[t][j] = e;
            sum += e;
        }
        float inv = 1.0f / sum;
        for (int j = 0; j < 64; ++j) sf[t][j] *= inv;
    }
    __syncthreads();

    // P -> f16 (zero rows >= 49 so padding cannot pollute the O WMMA)
    for (int idx = t; idx < 64 * 64; idx += 128) {
        int m = idx >> 6, n = idx & 63;
        ph[m][n] = (_Float16)(m < 49 ? sf[m][n] : 0.0f);
    }
    __syncthreads();

    // ---- O = P @ V ----
    v8f oacc[4];
    for (int i = 0; i < 4; ++i)
        for (int j = 0; j < 8; ++j) oacc[i][j] = 0.0f;
    for (int kk = 0; kk < 64; kk += 32) {
        const v16h a = make_afrag(&ph[rloc][0], kk + kbo);
#pragma unroll
        for (int nt = 0; nt < 4; ++nt) {
            const v16h b = *(const v16h*)&vt[(nt << 4) + (lane & 15)][kk + kbo2];
            oacc[nt] = __builtin_amdgcn_wmma_f32_16x16x32_f16(
                false, a, false, b, (short)0, oacc[nt], false, false);
        }
    }
#pragma unroll
    for (int nt = 0; nt < 4; ++nt) {
        int n = ncol + (nt << 4);
        if (n >= d) continue;
#pragma unroll
        for (int g = 0; g < 8; ++g) {
            int m = mrow + g;
            if (m >= 49) continue;
            O[(size_t)(w * 49 + m) * C + hd * d + n] = oacc[nt][g];
        }
    }
}

// ---------------------------------------------------------------------------
// Patch-merge gather: x [B, H*H, C] -> o [B, (H/2)^2, 4C]
// ---------------------------------------------------------------------------
__global__ void merge_gather(const float* __restrict__ x, float* __restrict__ o,
                             int Bn, int H, int C)
{
    int H2 = H / 2;
    int idx = blockIdx.x * 256 + threadIdx.x;
    int total = Bn * H2 * H2 * 4 * C;
    if (idx >= total) return;
    int cc = idx % (4 * C), t = idx / (4 * C);
    int b = t / (H2 * H2), rr = t % (H2 * H2);
    int r2 = rr / H2, c2 = rr % H2;
    int q = cc / C, c = cc % C;
    int dr = (q == 1 || q == 3) ? 1 : 0;
    int dc = (q >= 2) ? 1 : 0;
    o[idx] = x[((size_t)b * H * H + (2 * r2 + dr) * H + (2 * c2 + dc)) * C + c];
}

__global__ void mean_pool(const float* __restrict__ x, float* __restrict__ o,
                          int Bn, int L, int C)
{
    int idx = blockIdx.x * 256 + threadIdx.x;
    if (idx >= Bn * C) return;
    int b = idx / C, c = idx % C;
    float s = 0.0f;
    for (int l = 0; l < L; ++l) s += x[((size_t)b * L + l) * C + c];
    o[idx] = s / (float)L;
}

// ---------------------------------------------------------------------------
// Host orchestration
// ---------------------------------------------------------------------------
extern "C" void kernel_launch(void* const* d_in, const int* in_sizes, int n_in,
                              void* d_out, int out_size, void* d_ws, size_t ws_size,
                              hipStream_t stream)
{
    (void)in_sizes; (void)n_in; (void)out_size; (void)ws_size;
    const int Bn = 128;
    int pi = 0;
    auto nxt = [&]() { return (const float*)d_in[pi++]; };

    const float* X       = nxt();
    const float* patch_w = nxt();
    const float* patch_b = nxt();
    const float* eln_g   = nxt();
    const float* eln_b   = nxt();

    struct Blk { const float *ln1_g,*ln1_b,*wq,*bq,*wk,*bk,*wv,*bv,*wo,*bo,*bias,*ln2_g,*ln2_b,*w1,*b1,*w2,*b2; };
    struct Mrg { const float *g,*b,*w; };
    const int depths[3]   = {2, 3, 2};
    const int dims[3]     = {96, 192, 384};
    const int headsArr[3] = {3, 3, 6};
    const int resArr[3]   = {28, 14, 7};
    Blk blk[3][3];
    Mrg mrg[2];
    for (int s = 0; s < 3; ++s) {
        for (int i = 0; i < depths[s]; ++i) {
            Blk& p = blk[s][i];
            p.ln1_g = nxt(); p.ln1_b = nxt();
            p.wq = nxt(); p.bq = nxt();
            p.wk = nxt(); p.bk = nxt();
            p.wv = nxt(); p.bv = nxt();
            p.wo = nxt(); p.bo = nxt();
            p.bias = nxt();
            p.ln2_g = nxt(); p.ln2_b = nxt();
            p.w1 = nxt(); p.b1 = nxt();
            p.w2 = nxt(); p.b2 = nxt();
        }
        if (s < 2) { mrg[s].g = nxt(); mrg[s].b = nxt(); mrg[s].w = nxt(); }
    }
    const float* fln_g  = nxt();
    const float* fln_b  = nxt();
    const float* head_w = nxt();
    const float* head_b = nxt();

    // workspace layout (fp32 units): 6*U total; FFN intermediate aliases the
    // 4U span starting at bufB (bufB/Q/K/V are dead during the FFN).
    const size_t U = (size_t)128 * 784 * 96;  // 9,633,792 floats
    float* ws   = (float*)d_ws;
    float* h    = ws;
    float* bufA = ws + U;
    float* bufB = ws + 2 * U;
    float* bufQ = ws + 3 * U;
    float* bufK = ws + 4 * U;
    float* bufV = ws + 5 * U;
    float* bufM = bufB;

    auto gemm = [&](const float* A, const float* Bm, const float* bias, const float* R,
                    float* Cc, int M, int N, int K, int bT, int act) {
        if (!bT && (M % 128 == 0) && (K % 32 == 0) && (N % 4 == 0)) {
            dim3 g((N + 63) / 64, M / 128);
            gemm_f16wmma_fast<<<g, dim3(256), 0, stream>>>(A, Bm, bias, R, Cc, M, N, K, act);
        } else {
            dim3 g((N + 63) / 64, (M + 63) / 64);
            gemm_f16wmma_safe<<<g, dim3(128), 0, stream>>>(A, Bm, bias, R, Cc, M, N, K, bT, act);
        }
    };
    auto ln = [&](const float* x, const float* g, const float* b, float* y, int rows, int C) {
        layernorm_k<<<dim3((rows + 7) / 8), dim3(256), 0, stream>>>(x, g, b, y, rows, C);
    };

    // ---- patch embed: im2col -> GEMM (K=36, transposed weight) -> LN ----
    {
        int total = Bn * 784 * 36;
        im2col_patch<<<dim3((total + 255) / 256), dim3(256), 0, stream>>>(X, bufB, Bn);
        gemm(bufB, patch_w, patch_b, nullptr, bufA, Bn * 784, 96, 36, /*bT=*/1, 0);
        ln(bufA, eln_g, eln_b, h, Bn * 784, 96);
    }

    // ---- stages ----
    for (int s = 0; s < 3; ++s) {
        int H = resArr[s], C = dims[s], heads = headsArr[s];
        int M = Bn * H * H, d = C / heads;
        int shift_base = (H > 7) ? 3 : 0;
        for (int i = 0; i < depths[s]; ++i) {
            const Blk& p = blk[s][i];
            int shift = (i & 1) ? shift_base : 0;

            ln(h, p.ln1_g, p.ln1_b, bufA, M, C);
            {
                int total = M * C;
                win_part<<<dim3((total + 255) / 256), dim3(256), 0, stream>>>(bufA, bufB, Bn, H, C, shift);
            }
            gemm(bufB, p.wq, p.bq, nullptr, bufQ, M, C, C, 0, 0);
            gemm(bufB, p.wk, p.bk, nullptr, bufK, M, C, C, 0, 0);
            gemm(bufB, p.wv, p.bv, nullptr, bufV, M, C, C, 0, 0);

            int nWinTot = Bn * (H / 7) * (H / 7);
            win_attn<<<dim3(nWinTot * heads), dim3(128), 0, stream>>>(
                bufQ, bufK, bufV, p.bias, bufA, heads, d, C, H, shift);

            gemm(bufA, p.wo, p.bo, nullptr, bufB, M, C, C, 0, 0);
            {
                int total = M * C;
                win_rev_add<<<dim3((total + 255) / 256), dim3(256), 0, stream>>>(h, bufB, Bn, H, C, shift);
            }

            ln(h, p.ln2_g, p.ln2_b, bufA, M, C);
            gemm(bufA, p.w1, p.b1, nullptr, bufM, M, 4 * C, C, 0, /*gelu=*/1);
            gemm(bufM, p.w2, p.b2, /*R=*/h, h, M, C, 4 * C, 0, 0);
        }
        if (s < 2) {
            int H2 = H / 2, M2 = Bn * H2 * H2;
            {
                int total = M2 * 4 * C;
                merge_gather<<<dim3((total + 255) / 256), dim3(256), 0, stream>>>(h, bufA, Bn, H, C);
            }
            ln(bufA, mrg[s].g, mrg[s].b, bufB, M2, 4 * C);
            gemm(bufB, mrg[s].w, nullptr, nullptr, h, M2, 2 * C, 4 * C, 0, 0);
        }
    }

    // ---- final LN -> mean pool -> head ----
    ln(h, fln_g, fln_b, bufA, Bn * 49, 384);
    {
        int total = Bn * 384;
        mean_pool<<<dim3((total + 255) / 256), dim3(256), 0, stream>>>(bufA, bufB, Bn, 49, 384);
    }
    gemm(bufB, head_w, head_b, nullptr, (float*)d_out, Bn, 6, 384, 0, 0);
}
